// HourlyWaterQualityPredictor_5111011082893
// MI455X (gfx1250) — matrix operand order repaired and checked by reading, not compiled, not run on hardware
//
#include <hip/hip_runtime.h>
#include <hip/hip_bf16.h>

typedef __bf16 bf16;
typedef __attribute__((ext_vector_type(16))) __bf16 v16bf;
typedef __attribute__((ext_vector_type(8)))  float  v8f;

union Frag { uint4 u4[2]; v16bf v; };

__device__ __forceinline__ float sigmoidf_(float x) {
    return 1.0f / (1.0f + __expf(-x));
}
__device__ __forceinline__ float tanhf_(float x) {
    float ax = fabsf(x);
    float e  = __expf(-2.0f * ax);
    float r  = (1.0f - e) / (1.0f + e);
    return x < 0.0f ? -r : r;
}

// ---------------------------------------------------------------------------
// Weight pre-pack kernels: bf16, WMMA-B-fragment-native layout
//   out[(chunk*N + n)*32 + kk], kk = (e/8)*16 + kgroup*8 + e%8  <->  K = chunk*32 + kk
// ---------------------------------------------------------------------------

// conv_w [128oc][35ic][3][3] -> wpk [10][128][32]
//  chunks 0..8 : tap = chunk, kk = hidden channel (input ch 3+kk)
//  chunk  9    : kk -> (ic = kk/9 in 0..2, tap = kk%9), kk>=27 zero pad
//  n (output col) remapped: n = ch*4 + gate  (orig oc = gate*32 + ch)
__global__ void pack_conv(const float* __restrict__ w, bf16* __restrict__ out) {
    int idx = blockIdx.x * 256 + threadIdx.x;
    if (idx >= 10 * 128 * 32) return;
    int kk = idx & 31;
    int n  = (idx >> 5) & 127;
    int chunk = idx >> 12;
    int oc = (n & 3) * 32 + (n >> 2);
    float val = 0.f;
    if (chunk < 9) {
        int ky = chunk / 3, kx = chunk % 3;
        val = w[((oc * 35 + (3 + kk)) * 3 + ky) * 3 + kx];
    } else if (kk < 27) {
        int ic = kk / 9, tap = kk % 9;
        val = w[((oc * 35 + ic) * 3 + tap / 3) * 3 + tap % 3];
    }
    out[idx] = (bf16)val;
}

// generic linear: src [K][Nsrc] row-major -> out [K/32][N][32] (zero-pad cols)
__global__ void pack_lin(const float* __restrict__ w, bf16* __restrict__ out,
                         int K, int N, int Nsrc) {
    int idx = blockIdx.x * 256 + threadIdx.x;
    int total = (K >> 5) * N * 32;
    if (idx >= total) return;
    int kk = idx & 31;
    int n  = (idx >> 5) % N;
    int chunk = idx / (N * 32);
    int k = chunk * 32 + kk;
    float v = (n < Nsrc) ? w[k * Nsrc + n] : 0.f;
    out[idx] = (bf16)v;
}

// ---------------------------------------------------------------------------
// Fused ConvLSTM: 8 samples per WG (512 threads / 16 waves), 12 steps fused.
// M = 128 rows (8 samples x 16 positions), N = 128 gate-channels, K = 10x32.
// Wave w: N-tile w%8, M-tiles (w/8)*4 .. +3  ->  C[4] accumulators (32 VGPRs).
// Writes bf16 feature rows z0[B][544] (512 h features + 32 hour-embed).
// ---------------------------------------------------------------------------
#define SMP 8
#define NT  512

__global__ __launch_bounds__(NT) void convlstm_kernel(
    const float* __restrict__ x,      // [B,12,3,4,4]
    const float* __restrict__ hour,   // [B,1]
    const float* __restrict__ conv_b, // [128] (orig oc order)
    const float* __restrict__ he_w1, const float* __restrict__ he_b1,
    const float* __restrict__ he_w2, const float* __restrict__ he_b2,
    const bf16*  __restrict__ wpk,    // [10][128][32]
    bf16* __restrict__ z0)            // [B][544]
{
    __shared__ __align__(16) bf16  in_lds[SMP * 36 * 40]; // [s][6x6 pad pos][ch(40 stride)]
    __shared__ __align__(16) bf16  xim   [SMP * 16 * 40]; // [row][kk(40 stride)]  x im2col
    __shared__ float c_lds[SMP * 16 * 32];                // [s][p][ch] fp32 cell state
    __shared__ float x_stage[SMP * 3 * 16];               // [s][c][p]

    const int tid  = threadIdx.x;
    const int wg   = blockIdx.x;
    const int lane = tid & 31;
    const int w    = tid >> 5;       // wave id 0..15
    const int nt   = w & 7;          // N-tile
    const int mh   = w >> 3;         // M-half (0/1)
    const int lgrp = lane >> 4;
    const int lid  = lane & 15;

    // zero: h0 = 0, padded borders = 0, c0 = 0
    for (int i = tid; i < SMP * 36 * 40; i += NT) in_lds[i] = (bf16)0.f;
    for (int i = tid; i < SMP * 16 * 32; i += NT) c_lds[i] = 0.f;

    const int n    = nt * 16 + lid;         // packed output column
    const int gate = n & 3;                 // 0:i 1:f 2:o 3:g
    const int ch   = n >> 2;                // hidden channel 0..31
    const float bias = conv_b[gate * 32 + ch];
    const int py0 = lid >> 2, px0 = lid & 3;

    v8f C[4];

    for (int t = 0; t < 12; ++t) {
        // stage this timestep's x slab (coalesced)
        for (int i = tid; i < SMP * 48; i += NT) {
            int s = i / 48, cc = (i >> 4) % 3, pp = i & 15;
            x_stage[i] = x[(((size_t)(wg * SMP + s) * 12 + t) * 3 + cc) * 16 + pp];
        }
        __syncthreads();
        // build fragment-ready im2col of x: kk = ic*9 + tap (27 real + 5 zero)
        for (int i = tid; i < SMP * 16 * 32; i += NT) {
            int k = i & 31, pp = (i >> 5) & 15, s = i >> 9;
            float v = 0.f;
            if (k < 27) {
                int ic = k / 9, tap = k % 9;
                int yy = (pp >> 2) + tap / 3 - 1, xx = (pp & 3) + tap % 3 - 1;
                if (yy >= 0 && yy < 4 && xx >= 0 && xx < 4)
                    v = x_stage[(s * 3 + ic) * 16 + yy * 4 + xx];
            }
            xim[(s * 16 + pp) * 40 + k] = (bf16)v;
        }
        __syncthreads();

        // ---- GEMM: gates[128 rows][128 cols]; 4 tiles x 10 chunks per wave --
        #pragma unroll
        for (int m = 0; m < 4; ++m)
            #pragma unroll
            for (int i = 0; i < 8; ++i) C[m][i] = 0.f;

        for (int chunk = 0; chunk < 10; ++chunk) {
            Frag bfr;
            const bf16* bsrc = wpk + (size_t)(chunk * 128 + n) * 32 + lgrp * 8;
            bfr.u4[0] = *(const uint4*)(bsrc);
            bfr.u4[1] = *(const uint4*)(bsrc + 16);
            if (chunk < 9) {
                const int off = ((py0 + chunk / 3) * 6 + (px0 + chunk % 3)) * 40 + lgrp * 8;
                #pragma unroll
                for (int mi = 0; mi < 4; ++mi) {
                    Frag af;
                    const bf16* asrc = in_lds + (mh * 4 + mi) * 1440 + off;
                    af.u4[0] = *(const uint4*)(asrc);
                    af.u4[1] = *(const uint4*)(asrc + 16);
                    C[mi] = __builtin_amdgcn_wmma_f32_16x16x32_bf16(
                        false, af.v, false, bfr.v, (short)0, C[mi], false, false);
                }
            } else {
                #pragma unroll
                for (int mi = 0; mi < 4; ++mi) {
                    Frag af;
                    const bf16* asrc = xim + ((mh * 4 + mi) * 16 + lid) * 40 + lgrp * 8;
                    af.u4[0] = *(const uint4*)(asrc);
                    af.u4[1] = *(const uint4*)(asrc + 16);
                    C[mi] = __builtin_amdgcn_wmma_f32_16x16x32_bf16(
                        false, af.v, false, bfr.v, (short)0, C[mi], false, false);
                }
            }
        }
        __syncthreads();   // all GEMM reads of in_lds done

        // ---- LSTM pointwise update: gather 4 gates via quad shuffles ----
        for (int si = 0; si < 4; ++si) {
            const int s = mh * 4 + si;
            #pragma unroll
            for (int r = 0; r < 8; ++r) {
                float v  = C[si][r] + bias;
                float x1 = __shfl_xor(v, 1);
                float x2 = __shfl_xor(v, 2);
                float x3 = __shfl_xor(v, 3);
                float g4[4];
                g4[gate]     = v;
                g4[gate ^ 1] = x1;
                g4[gate ^ 2] = x2;
                g4[gate ^ 3] = x3;
                int pu = r + 8 * lgrp;
                float cold = c_lds[(s * 16 + pu) * 32 + ch];
                float cn = sigmoidf_(g4[1]) * cold + sigmoidf_(g4[0]) * tanhf_(g4[3]);
                float hn = sigmoidf_(g4[2]) * tanhf_(cn);
                if (gate == 0) {
                    c_lds[(s * 16 + pu) * 32 + ch] = cn;
                    in_lds[s * 1440 + (((pu >> 2) + 1) * 6 + (pu & 3) + 1) * 40 + ch] = (bf16)hn;
                }
            }
        }
        __syncthreads();
    }

    // ---- write bf16 feature rows: 512 h features (ch*16 + pos order) ----
    for (int i = tid; i < SMP * 512; i += NT) {
        int s = i >> 9, f = i & 511;
        int cc = f >> 4, pp = f & 15;
        z0[(size_t)(wg * SMP + s) * 544 + f] =
            in_lds[s * 1440 + (((pp >> 2) + 1) * 6 + (pp & 3) + 1) * 40 + cc];
    }
    // hour embedding -> features 512..543
    if (tid < SMP) {
        int s = tid;
        float hr = hour[wg * SMP + s];
        float t1[16];
        #pragma unroll
        for (int j = 0; j < 16; ++j) t1[j] = fmaxf(hr * he_w1[j] + he_b1[j], 0.f);
        for (int k = 0; k < 32; ++k) {
            float acc = he_b2[k];
            #pragma unroll
            for (int j = 0; j < 16; ++j) acc += t1[j] * he_w2[j * 32 + k];
            z0[(size_t)(wg * SMP + s) * 544 + 512 + k] = (bf16)acc;
        }
    }
}

// ---------------------------------------------------------------------------
// Decoder: 64 rows/WG, 512 threads / 16 waves.
// 544->256 (relu) -> 128 (relu) -> 30 (sigmoid)
// ---------------------------------------------------------------------------
__global__ __launch_bounds__(NT) void decoder_kernel(
    const bf16* __restrict__ z0,   // [B][544]
    const bf16* __restrict__ w1pk, // [17][256][32]
    const bf16* __restrict__ w2pk, // [8][128][32]
    const bf16* __restrict__ w3pk, // [4][32][32]
    const float* __restrict__ b1, const float* __restrict__ b2,
    const float* __restrict__ b3,
    float* __restrict__ out)       // [B][30]
{
    __shared__ __align__(16) bf16 z1[64 * 264];  // padded stride
    __shared__ __align__(16) bf16 z2[64 * 136];

    const int tid  = threadIdx.x;
    const int lane = tid & 31;
    const int w    = tid >> 5;      // 0..15
    const int lgrp = lane >> 4;
    const int lid  = lane & 15;
    const int base = blockIdx.x * 64;

    // ---- layer 1: M=64 N=256 K=544 ; wave w -> n-tile w, m 0..3 ----
    {
        v8f C[4];
        #pragma unroll
        for (int m = 0; m < 4; ++m)
            #pragma unroll
            for (int i = 0; i < 8; ++i) C[m][i] = 0.f;

        for (int chunk = 0; chunk < 17; ++chunk) {
            Frag B0;
            const bf16* p0 = w1pk + (size_t)(chunk * 256 + w * 16 + lid) * 32 + lgrp * 8;
            B0.u4[0] = *(const uint4*)(p0);  B0.u4[1] = *(const uint4*)(p0 + 16);
            #pragma unroll
            for (int m = 0; m < 4; ++m) {
                Frag A;
                const bf16* asrc = z0 + (size_t)(base + m * 16 + lid) * 544 + chunk * 32 + lgrp * 8;
                A.u4[0] = *(const uint4*)(asrc);
                A.u4[1] = *(const uint4*)(asrc + 16);
                C[m] = __builtin_amdgcn_wmma_f32_16x16x32_bf16(
                    false, A.v, false, B0.v, (short)0, C[m], false, false);
            }
        }
        int nn = w * 16 + lid;
        float bb = b1[nn];
        #pragma unroll
        for (int m = 0; m < 4; ++m)
            #pragma unroll
            for (int r = 0; r < 8; ++r)
                z1[(m * 16 + r + 8 * lgrp) * 264 + nn] = (bf16)fmaxf(C[m][r] + bb, 0.f);
    }
    __syncthreads();

    // ---- layer 2: M=64 N=128 K=256 ; wave w -> n-tile w%8, m-half w/8 ----
    {
        v8f C[2];
        #pragma unroll
        for (int m = 0; m < 2; ++m)
            #pragma unroll
            for (int i = 0; i < 8; ++i) C[m][i] = 0.f;

        const int nt2 = w & 7, mh2 = w >> 3;
        for (int chunk = 0; chunk < 8; ++chunk) {
            Frag B0;
            const bf16* p0 = w2pk + (size_t)(chunk * 128 + nt2 * 16 + lid) * 32 + lgrp * 8;
            B0.u4[0] = *(const uint4*)(p0);  B0.u4[1] = *(const uint4*)(p0 + 16);
            #pragma unroll
            for (int mi = 0; mi < 2; ++mi) {
                Frag A;
                const bf16* asrc = z1 + ((mh2 * 2 + mi) * 16 + lid) * 264 + chunk * 32 + lgrp * 8;
                A.u4[0] = *(const uint4*)(asrc);
                A.u4[1] = *(const uint4*)(asrc + 16);
                C[mi] = __builtin_amdgcn_wmma_f32_16x16x32_bf16(
                    false, A.v, false, B0.v, (short)0, C[mi], false, false);
            }
        }
        int nn = (w & 7) * 16 + lid;
        float bb = b2[nn];
        #pragma unroll
        for (int mi = 0; mi < 2; ++mi)
            #pragma unroll
            for (int r = 0; r < 8; ++r)
                z2[(((w >> 3) * 2 + mi) * 16 + r + 8 * lgrp) * 136 + nn] =
                    (bf16)fmaxf(C[mi][r] + bb, 0.f);
    }
    __syncthreads();

    // ---- layer 3: M=64 N=32(30) K=128 ; waves 0..7: (m = w/2, ntile = w&1) --
    if (w < 8) {
        v8f C;
        #pragma unroll
        for (int i = 0; i < 8; ++i) C[i] = 0.f;
        int m = w >> 1, nt3 = w & 1;
        for (int chunk = 0; chunk < 4; ++chunk) {
            Frag B0, A;
            const bf16* p0 = w3pk + (size_t)(chunk * 32 + nt3 * 16 + lid) * 32 + lgrp * 8;
            B0.u4[0] = *(const uint4*)(p0);  B0.u4[1] = *(const uint4*)(p0 + 16);
            const bf16* asrc = z2 + (m * 16 + lid) * 136 + chunk * 32 + lgrp * 8;
            A.u4[0] = *(const uint4*)(asrc);
            A.u4[1] = *(const uint4*)(asrc + 16);
            C = __builtin_amdgcn_wmma_f32_16x16x32_bf16(
                false, A.v, false, B0.v, (short)0, C, false, false);
        }
        int nn = nt3 * 16 + lid;
        float bb = (nn < 30) ? b3[nn] : 0.f;
        #pragma unroll
        for (int r = 0; r < 8; ++r) {
            float v = sigmoidf_(C[r] + bb);
            int row = base + m * 16 + r + 8 * lgrp;
            if (nn < 30) out[(size_t)row * 30 + nn] = v;
        }
    }
}

// ---------------------------------------------------------------------------
extern "C" void kernel_launch(void* const* d_in, const int* in_sizes, int n_in,
                              void* d_out, int out_size, void* d_ws, size_t ws_size,
                              hipStream_t stream) {
    (void)n_in; (void)out_size; (void)ws_size;
    const float* x      = (const float*)d_in[0];
    const float* hour   = (const float*)d_in[1];
    const float* conv_w = (const float*)d_in[2];
    const float* conv_b = (const float*)d_in[3];
    const float* he_w1  = (const float*)d_in[4];
    const float* he_b1  = (const float*)d_in[5];
    const float* he_w2  = (const float*)d_in[6];
    const float* he_b2  = (const float*)d_in[7];
    const float* d_w1   = (const float*)d_in[8];
    const float* d_b1   = (const float*)d_in[9];
    const float* d_w2   = (const float*)d_in[10];
    const float* d_b2   = (const float*)d_in[11];
    const float* d_w3   = (const float*)d_in[12];
    const float* d_b3   = (const float*)d_in[13];

    const int B = in_sizes[0] / (12 * 3 * 16);

    // workspace layout (bytes, all 16B aligned)
    char* ws = (char*)d_ws;
    bf16* wpk  = (bf16*)(ws);                      // 10*128*32*2  =  81,920
    bf16* w1pk = (bf16*)(ws + 81920);              // 17*256*32*2  = 278,528
    bf16* w2pk = (bf16*)(ws + 81920 + 278528);     //  8*128*32*2  =  65,536
    bf16* w3pk = (bf16*)(ws + 81920 + 278528 + 16384);          // 8,192
    bf16* z0   = (bf16*)(ws + 81920 + 278528 + 16384 + 8192);   // B*544*2

    pack_conv<<<(10 * 128 * 32 + 255) / 256, 256, 0, stream>>>(conv_w, wpk);
    pack_lin<<<(17 * 256 * 32 + 255) / 256, 256, 0, stream>>>(d_w1, w1pk, 544, 256, 256);
    pack_lin<<<( 8 * 128 * 32 + 255) / 256, 256, 0, stream>>>(d_w2, w2pk, 256, 128, 128);
    pack_lin<<<( 4 *  32 * 32 + 255) / 256, 256, 0, stream>>>(d_w3, w3pk, 128, 32, 30);

    convlstm_kernel<<<B / 8, NT, 0, stream>>>(x, hour, conv_b,
                                              he_w1, he_b1, he_w2, he_b2,
                                              wpk, z0);
    decoder_kernel<<<B / 64, NT, 0, stream>>>(z0, w1pk, w2pk, w3pk,
                                              d_b1, d_b2, d_b3, (float*)d_out);
}
